// MultiHeadAttention_24421184045057
// MI455X (gfx1250) — matrix operand order repaired and checked by reading, not compiled, not run on hardware
//
#include <hip/hip_runtime.h>

// Problem constants (from reference)
#define Bv  2
#define Sv  2048
#define Dv  1024
#define Hv  16
#define DHv 64

typedef __attribute__((ext_vector_type(16))) __bf16 v16bf;
typedef __attribute__((ext_vector_type(8)))  __bf16 v8bf;
typedef __attribute__((ext_vector_type(8)))  float  v8f;

// ---------------------------------------------------------------------------
// WMMA fragment helpers (layouts per CDNA5 ISA 7.12.2, wave32)
// ---------------------------------------------------------------------------

// A fragment 16x32 (M x K), source row-major fp32 with leading dim ld.
// lane<16: M=lane, halves 0..7 -> K=0..7, halves 8..15 -> K=16..23
// lane>=16: M=lane-16, K-base +8.
__device__ __forceinline__ v16bf frag_a_f32(const float* src, int ld, int row0, int k0) {
  int lane = threadIdx.x & 31;
  int m    = lane & 15;
  int kb   = (lane >> 4) << 3;          // 0 or 8
  const float* p = src + (size_t)(row0 + m) * ld + k0 + kb;
  v16bf a;
#pragma unroll
  for (int i = 0; i < 8; ++i) a[i] = (__bf16)p[i];
#pragma unroll
  for (int i = 0; i < 8; ++i) a[8 + i] = (__bf16)p[16 + i];
  return a;
}

// Same A fragment but source is bf16 (two 16B vector loads).
__device__ __forceinline__ v16bf frag_a_bf16(const __bf16* src, int ld, int row0, int k0) {
  int lane = threadIdx.x & 31;
  int m    = lane & 15;
  int kb   = (lane >> 4) << 3;
  const __bf16* p = src + (size_t)(row0 + m) * ld + k0 + kb;
  v8bf lo = *(const v8bf*)p;
  v8bf hi = *(const v8bf*)(p + 16);
  v16bf a;
#pragma unroll
  for (int i = 0; i < 8; ++i) { a[i] = lo[i]; a[8 + i] = hi[i]; }
  return a;
}

// B fragment 32x16 (K x N) from an N-major bf16 array: B[k][n] = src[(n0+n)*ld + k0 + k].
// lane<16: K=0..7,16..23, lane>=16: K=8..15,24..31; N = lane&15.
__device__ __forceinline__ v16bf frag_b_nmajor(const __bf16* src, int ld, int n0, int k0) {
  int lane = threadIdx.x & 31;
  int n    = lane & 15;
  int kb   = (lane >> 4) << 3;          // 0 or 8
  const __bf16* p = src + (size_t)(n0 + n) * ld + k0 + kb;
  v8bf lo = *(const v8bf*)p;
  v8bf hi = *(const v8bf*)(p + 16);
  v16bf b;
#pragma unroll
  for (int i = 0; i < 8; ++i) { b[i] = lo[i]; b[8 + i] = hi[i]; }
  return b;
}

__device__ __forceinline__ v8f wmma_bf16(v16bf a, v16bf b, v8f c) {
  return __builtin_amdgcn_wmma_f32_16x16x32_bf16(false, a, false, b, (short)0, c, false, false);
}

// ---------------------------------------------------------------------------
// K0: fp32 weight (D,D) -> bf16 transposed (N-major): wt[j*D+i] = w[i*D+j]
// ---------------------------------------------------------------------------
__global__ void mha_wconv(const float* __restrict__ w, __bf16* __restrict__ wt) {
  int tid = blockIdx.x * 256 + threadIdx.x;   // over D*D
  int i = tid / Dv, j = tid % Dv;
  wt[(size_t)j * Dv + i] = (__bf16)w[(size_t)i * Dv + j];
}

// ---------------------------------------------------------------------------
// K1: projection GEMM  Y = X(4096x1024 fp32) @ W(bf16,N-major) + bias
//     transpose_out==0 -> out[b][h][s][dh]   (Q, K)
//     transpose_out==1 -> out[b][h][dh][s]   (V, pre-transposed for PV GEMM)
// grid 512 blocks x 256 threads; wave computes 16x64 tile, K-loop 1024.
// ---------------------------------------------------------------------------
__global__ void mha_proj(const float* __restrict__ x, const __bf16* __restrict__ wt,
                         const float* __restrict__ bias, __bf16* __restrict__ out,
                         int transpose_out) {
  int wave  = threadIdx.x >> 5;
  int mtile = blockIdx.x >> 1;
  int ngrp  = blockIdx.x & 1;
  int row0  = mtile * 16;
  int n0    = ngrp * 512 + wave * 64;

  v8f acc[4] = {v8f{}, v8f{}, v8f{}, v8f{}};
  for (int k0 = 0; k0 < Dv; k0 += 32) {
    v16bf a = frag_a_f32(x, Dv, row0, k0);
#pragma unroll
    for (int t = 0; t < 4; ++t) {
      v16bf b = frag_b_nmajor(wt, Dv, n0 + t * 16, k0);
      acc[t] = wmma_bf16(a, b, acc[t]);
    }
  }

  int lane = threadIdx.x & 31;
  int n_l  = lane & 15;
  int mb   = (lane >> 4) << 3;
#pragma unroll
  for (int t = 0; t < 4; ++t) {
#pragma unroll
    for (int r = 0; r < 8; ++r) {
      int m  = row0 + mb + r;
      int nn = n0 + t * 16 + n_l;
      float v = acc[t][r] + bias[nn];
      int b = m / Sv, s = m % Sv, h = nn / DHv, dh = nn % DHv;
      size_t idx = transpose_out
        ? (((size_t)b * Hv + h) * DHv + dh) * Sv + s
        : (((size_t)b * Hv + h) * Sv + s) * DHv + dh;
      out[idx] = (__bf16)v;
    }
  }
}

// ---------------------------------------------------------------------------
// K2: scores (QK^T/8, causal) + softmax, writes fp32 attn rows to d_out.
// One block per (b,h, 16 q-rows). 128KB dynamic LDS holds the 16x2048 strip.
// ---------------------------------------------------------------------------
extern __shared__ float mha_smem[];

__global__ void mha_scores_softmax(const __bf16* __restrict__ Qh,
                                   const __bf16* __restrict__ Kh,
                                   float* __restrict__ attn) {
  int bh = blockIdx.x >> 7;       // 0..31
  int qt = blockIdx.x & 127;
  int q0 = qt * 16;
  const __bf16* Q = Qh + (size_t)bh * Sv * DHv;
  const __bf16* K = Kh + (size_t)bh * Sv * DHv;
  float (*lds)[Sv] = (float (*)[Sv])mha_smem;

  int wave = threadIdx.x >> 5;
  int lane = threadIdx.x & 31;
  int n_l  = lane & 15;
  int mb   = (lane >> 4) << 3;

  // wave w owns columns [256w, 256w+256); skip fully-masked tiles (wave-uniform)
  for (int nt = 0; nt < 16; ++nt) {
    int n0 = wave * 256 + nt * 16;
    if (n0 > q0 + 15) break;
    v8f acc = v8f{};
#pragma unroll
    for (int k0 = 0; k0 < DHv; k0 += 32) {
      v16bf a = frag_a_bf16(Q, DHv, q0, k0);
      v16bf b = frag_b_nmajor(K, DHv, n0, k0);
      acc = wmma_bf16(a, b, acc);
    }
#pragma unroll
    for (int r = 0; r < 8; ++r)
      lds[mb + r][n0 + n_l] = acc[r] * 0.125f;   // 1/sqrt(DH)
  }
  __syncthreads();

  // exact softmax: 16 threads per row, causal length L = qi+1
  int r  = threadIdx.x >> 4;
  int c  = threadIdx.x & 15;
  int qi = q0 + r;
  int L  = qi + 1;

  float mx = -3.0e38f;
  for (int j = c; j < L; j += 16) mx = fmaxf(mx, lds[r][j]);
#pragma unroll
  for (int off = 8; off; off >>= 1) mx = fmaxf(mx, __shfl_xor(mx, off, 16));

  float sum = 0.f;
  for (int j = c; j < L; j += 16) {
    float e = __expf(lds[r][j] - mx);
    lds[r][j] = e;
    sum += e;
  }
#pragma unroll
  for (int off = 8; off; off >>= 1) sum += __shfl_xor(sum, off, 16);
  float inv = 1.f / sum;

  float* arow = attn + ((size_t)bh * Sv + qi) * Sv;
  for (int j = c; j < Sv; j += 16) arow[j] = (j < L) ? lds[r][j] * inv : 0.f;
}

// ---------------------------------------------------------------------------
// K3: ctx = attn @ V  (attn fp32 read back from d_out, V pre-transposed bf16)
// Wave computes 16 q-rows x 64 (DH); K-loop clipped at causal boundary.
// ctx stored bf16 in [B, S, H*DH] (the transpose+reshape of the reference).
// ---------------------------------------------------------------------------
__global__ void mha_ctx(const float* __restrict__ attn, const __bf16* __restrict__ Vt,
                        __bf16* __restrict__ ctx) {
  int bh    = blockIdx.x >> 4;    // 0..31
  int chunk = blockIdx.x & 15;
  int wave  = threadIdx.x >> 5;
  int q0    = chunk * 128 + wave * 16;
  const float*  A = attn + (size_t)bh * Sv * Sv;
  const __bf16* V = Vt   + (size_t)bh * DHv * Sv;

  v8f acc[4] = {v8f{}, v8f{}, v8f{}, v8f{}};
  int kmax = q0 + 16;             // attn is exactly 0 beyond the causal edge
  for (int k0 = 0; k0 < kmax; k0 += 32) {
    v16bf a = frag_a_f32(A, Sv, q0, k0);
#pragma unroll
    for (int t = 0; t < 4; ++t) {
      v16bf b = frag_b_nmajor(V, Sv, t * 16, k0);
      acc[t] = wmma_bf16(a, b, acc[t]);
    }
  }

  int b = bh / Hv, h = bh % Hv;
  int lane = threadIdx.x & 31;
  int n_l  = lane & 15;
  int mb   = (lane >> 4) << 3;
#pragma unroll
  for (int t = 0; t < 4; ++t) {
#pragma unroll
    for (int r = 0; r < 8; ++r) {
      int qi = q0 + mb + r;
      int d  = t * 16 + n_l;
      ctx[((size_t)b * Sv + qi) * Dv + h * DHv + d] = (__bf16)acc[t][r];
    }
  }
}

// ---------------------------------------------------------------------------
// K4: out = ctx @ wo + bo   (4096x1024x1024), fp32 output to d_out
// ---------------------------------------------------------------------------
__global__ void mha_outproj(const __bf16* __restrict__ ctx, const __bf16* __restrict__ wt,
                            const float* __restrict__ bias, float* __restrict__ out) {
  int wave  = threadIdx.x >> 5;
  int mtile = blockIdx.x >> 1;
  int ngrp  = blockIdx.x & 1;
  int row0  = mtile * 16;
  int n0    = ngrp * 512 + wave * 64;

  v8f acc[4] = {v8f{}, v8f{}, v8f{}, v8f{}};
  for (int k0 = 0; k0 < Dv; k0 += 32) {
    v16bf a = frag_a_bf16(ctx, Dv, row0, k0);
#pragma unroll
    for (int t = 0; t < 4; ++t) {
      v16bf b = frag_b_nmajor(wt, Dv, n0 + t * 16, k0);
      acc[t] = wmma_bf16(a, b, acc[t]);
    }
  }

  int lane = threadIdx.x & 31;
  int n_l  = lane & 15;
  int mb   = (lane >> 4) << 3;
#pragma unroll
  for (int t = 0; t < 4; ++t) {
#pragma unroll
    for (int r = 0; r < 8; ++r) {
      int m  = row0 + mb + r;
      int nn = n0 + t * 16 + n_l;
      out[(size_t)m * Dv + nn] = acc[t][r] + bias[nn];
    }
  }
}

// ---------------------------------------------------------------------------
// Launch
// ---------------------------------------------------------------------------
extern "C" void kernel_launch(void* const* d_in, const int* in_sizes, int n_in,
                              void* d_out, int out_size, void* d_ws, size_t ws_size,
                              hipStream_t stream) {
  (void)in_sizes; (void)n_in; (void)out_size; (void)ws_size;
  const float* q    = (const float*)d_in[0];
  const float* k    = (const float*)d_in[1];
  const float* v    = (const float*)d_in[2];
  // d_in[3] = mask (causal, hardcoded)
  const float* wq   = (const float*)d_in[4];
  const float* bq   = (const float*)d_in[5];
  const float* wk   = (const float*)d_in[6];
  const float* bk   = (const float*)d_in[7];
  const float* wv   = (const float*)d_in[8];
  const float* bv   = (const float*)d_in[9];
  const float* wo   = (const float*)d_in[10];
  const float* bo   = (const float*)d_in[11];

  // Workspace carve-up (bf16 elements): 4 weights + Qh + Kh + Vt + ctx = 40 MB
  __bf16* wq_t = (__bf16*)d_ws;
  __bf16* wk_t = wq_t + (size_t)Dv * Dv;
  __bf16* wv_t = wk_t + (size_t)Dv * Dv;
  __bf16* wo_t = wv_t + (size_t)Dv * Dv;
  __bf16* Qh   = wo_t + (size_t)Dv * Dv;                 // [B,H,S,DH]
  __bf16* Kh   = Qh   + (size_t)Bv * Hv * Sv * DHv;      // [B,H,S,DH]
  __bf16* Vt   = Kh   + (size_t)Bv * Hv * Sv * DHv;      // [B,H,DH,S]
  __bf16* ctx  = Vt   + (size_t)Bv * Hv * Sv * DHv;      // [B,S,D]

  float* out  = (float*)d_out;
  float* attn = out + (size_t)Bv * Sv * Dv;              // [B,H,S,S]

  dim3 blk(256);

  mha_wconv<<<Dv * Dv / 256, blk, 0, stream>>>(wq, wq_t);
  mha_wconv<<<Dv * Dv / 256, blk, 0, stream>>>(wk, wk_t);
  mha_wconv<<<Dv * Dv / 256, blk, 0, stream>>>(wv, wv_t);
  mha_wconv<<<Dv * Dv / 256, blk, 0, stream>>>(wo, wo_t);

  mha_proj<<<512, blk, 0, stream>>>(q, wq_t, bq, Qh, 0);
  mha_proj<<<512, blk, 0, stream>>>(k, wk_t, bk, Kh, 0);
  mha_proj<<<512, blk, 0, stream>>>(v, wv_t, bv, Vt, 1);

  mha_scores_softmax<<<Bv * Hv * (Sv / 16), blk, 16 * Sv * sizeof(float), stream>>>(Qh, Kh, attn);

  mha_ctx<<<Bv * Hv * (Sv / 128), blk, 0, stream>>>(attn, Vt, ctx);

  mha_outproj<<<512, blk, 0, stream>>>(ctx, wo_t, bo, out);
}
